// CustomConvLayer_24558622999210
// MI455X (gfx1250) — compile-verified, matrix operand order checked by reading, not run on hardware
//
#include <hip/hip_runtime.h>

typedef __attribute__((ext_vector_type(16))) _Float16 v16h;
typedef __attribute__((ext_vector_type(8)))  _Float16 v8h;
typedef __attribute__((ext_vector_type(8)))  float    v8f;

// Half-slot -> K mapping for 16-bit WMMA fragments (16x32 A / 32x16 B), wave32:
// lanes 0-15 : halfs 0..7 -> K 0..7,  halfs 8..15 -> K 16..23
// lanes 16-31: halfs 0..7 -> K 8..15, halfs 8..15 -> K 24..31
__device__ __host__ __forceinline__ int frag_k(int lane, int s) {
  return s + (s & 8) + ((lane & 16) >> 1);
}

// ---------------------------------------------------------------------------
// Implicit-GEMM 3x3 conv, v_wmma_f32_16x16x32_f16, zero LDS in the hot loop.
//   Input : NHWC f16, pre-padded [B][Hp][Wp][Ci]  (K ordered tap-major: tap*Ci+ci)
//   Weight: fragment-shuffled f16  [co_tile][kstep][lane][16]
//   Wave = 16 positions x 64 out channels (4 accumulators, 4 WMMA / K-step).
//   Workgroup = 8 waves = 128 positions x 64 channels.
//   f32 NCHW output goes through an LDS C-tile transpose for coalesced stores.
// ---------------------------------------------------------------------------
__global__ __launch_bounds__(256) void conv_wmma_kernel(
    const _Float16* __restrict__ xh, const _Float16* __restrict__ wsh,
    const float* __restrict__ bias, void* __restrict__ outp,
    int Ci, int ciShift, int Hp, int Wp, int CoT, int Ho, int Wo,
    int stride, long wstride, float slope, int out_f32_nchw)
{
  __shared__ float ctile[64][132];   // used only on the f32-NCHW path
  const int npos = Ho * Wo;
  const int nk   = (Ci * 9) >> 5;
  const int bb   = blockIdx.z;
  const int co_base = blockIdx.y * 64;
  const int tid  = threadIdx.x, lane = tid & 31, wv = tid >> 5;
  const int posb = (blockIdx.x * 8 + wv) * 16;

  // A row for this lane (rows striped over lanes&15; K split over lane halves)
  int m_row = posb + (lane & 15);
  int mc    = m_row < npos ? m_row : npos - 1;
  int oh = mc / Wo, ow = mc - oh * Wo;
  const _Float16* arow = xh + (((long)bb * Hp + oh * stride) * Wp + ow * stride) * Ci
                            + ((lane & 16) >> 1);
  const _Float16* wb = wsh + (long)bb * wstride
                            + ((long)(co_base >> 4) * nk) * 512 + lane * 16;

  v8f acc0 = {}, acc1 = {}, acc2 = {}, acc3 = {};
  for (int ks = 0; ks < nk; ++ks) {
    const int k0  = ks << 5;                 // uniform (SALU)
    const int tap = k0 >> ciShift;
    const int ci0 = k0 & (Ci - 1);
    const int kh  = tap / 3, kw = tap - kh * 3;
    const _Float16* ap = arow + ((long)kh * Wp + kw) * Ci + ci0;
    v8h lo = *(const v8h*)ap;                // K = k0+off .. +7
    v8h hi = *(const v8h*)(ap + 16);         // K = k0+16+off .. +7
    v16h a = __builtin_shufflevector(lo, hi, 0,1,2,3,4,5,6,7,8,9,10,11,12,13,14,15);

    const _Float16* bp = wb + (long)ks * 512;
    v16h b0 = *(const v16h*)(bp);
    v16h b1 = *(const v16h*)(bp + (long)nk * 512);
    v16h b2 = *(const v16h*)(bp + (long)nk * 512 * 2);
    v16h b3 = *(const v16h*)(bp + (long)nk * 512 * 3);
    if (ks + 1 < nk) __builtin_prefetch(bp + 512, 0, 1);

    acc0 = __builtin_amdgcn_wmma_f32_16x16x32_f16(false, a, false, b0, (short)0, acc0, false, false);
    acc1 = __builtin_amdgcn_wmma_f32_16x16x32_f16(false, a, false, b1, (short)0, acc1, false, false);
    acc2 = __builtin_amdgcn_wmma_f32_16x16x32_f16(false, a, false, b2, (short)0, acc2, false, false);
    acc3 = __builtin_amdgcn_wmma_f32_16x16x32_f16(false, a, false, b3, (short)0, acc3, false, false);
  }

  // C layout: VGPR r <-> M=r (lanes 0-15) / M=r+8 (lanes 16-31); N = lane&15
  if (out_f32_nchw) {
    // Stage the 64ch x 128pos workgroup C-tile to LDS, then write each channel
    // row as contiguous 128B chunks (coalesced NCHW f32 stores).
    #pragma unroll
    for (int r = 0; r < 8; ++r) {
      int ml = (wv << 4) + r + ((lane & 16) >> 1);
      int nl = lane & 15;
      ctile[nl     ][ml] = acc0[r];
      ctile[nl + 16][ml] = acc1[r];
      ctile[nl + 32][ml] = acc2[r];
      ctile[nl + 48][ml] = acc3[r];
    }
    __syncthreads();
    int n  = tid >> 2;                 // 0..63
    int ms = (tid & 3) << 5;           // 0,32,64,96
    int nG = co_base + n;
    float bi = bias ? bias[nG] : 0.f;
    int m0 = blockIdx.x * 128 + ms;
    float* orow = (float*)outp + ((long)bb * CoT + nG) * npos + m0;
    #pragma unroll
    for (int j = 0; j < 32; ++j) {
      if (m0 + j < npos) {
        float v = ctile[n][ms + j] + bi;
        v = (v >= 0.f) ? v : v * slope;
        orow[j] = v;
      }
    }
  } else {
    #pragma unroll
    for (int r = 0; r < 8; ++r) {
      int m = posb + r + ((lane & 16) >> 1);
      if (m >= npos) continue;
      float v0 = acc0[r], v1 = acc1[r], v2 = acc2[r], v3 = acc3[r];
      int n0 = co_base + (lane & 15);
      if (bias) { v0 += bias[n0]; v1 += bias[n0+16]; v2 += bias[n0+32]; v3 += bias[n0+48]; }
      v0 = (v0 >= 0.f) ? v0 : v0 * slope;
      v1 = (v1 >= 0.f) ? v1 : v1 * slope;
      v2 = (v2 >= 0.f) ? v2 : v2 * slope;
      v3 = (v3 >= 0.f) ? v3 : v3 * slope;
      _Float16* o = (_Float16*)outp + ((long)bb * npos + m) * CoT + n0;
      o[0]  = (_Float16)v0;
      o[16] = (_Float16)v1;
      o[32] = (_Float16)v2;
      o[48] = (_Float16)v3;
    }
  }
}

// ---------------------------------------------------------------------------
// Watermark MLP: coff[32][128] = (lrelu(wm@W1^T + b1, 0.2))@W2^T + b2
// ---------------------------------------------------------------------------
__global__ void mlp_coff_kernel(const float* __restrict__ wm,
                                const float* __restrict__ w1, const float* __restrict__ b1,
                                const float* __restrict__ w2, const float* __restrict__ b2,
                                float* __restrict__ coff)
{
  __shared__ float sh_in[32];
  __shared__ float sh_h[128];
  int t = threadIdx.x; // 128 threads
  for (int b = 0; b < 32; ++b) {
    if (t < 32) sh_in[t] = wm[b * 32 + t];
    __syncthreads();
    float h = b1[t];
    for (int i = 0; i < 32; ++i) h += sh_in[i] * w1[t * 32 + i];
    h = (h >= 0.f) ? h : 0.2f * h;
    sh_h[t] = h;
    __syncthreads();
    float o = b2[t];
    for (int k = 0; k < 128; ++k) o += sh_h[k] * w2[t * 128 + k];
    coff[b * 128 + t] = o;
    __syncthreads();
  }
}

// NCHW f32 -> NHWC f16 with coff scale + 2x2 avg pool. LDS tile transpose so
// both global reads (along w) and writes (along c) are coalesced.
// grid (8 ctiles, 64 h, 32 b), 256 threads
__global__ __launch_bounds__(256) void scale_pool_kernel(
    const float* __restrict__ x, const float* __restrict__ coff,
    _Float16* __restrict__ t0h)
{
  __shared__ _Float16 tile[16][64];
  int c0 = blockIdx.x * 16, h = blockIdx.y, b = blockIdx.z;
  int t = threadIdx.x;
  #pragma unroll
  for (int it = 0; it < 4; ++it) {
    int ci = it * 4 + (t >> 6);
    int w  = t & 63;
    const float* xp = x + (((long)b * 128 + c0 + ci) * 128 + h * 2) * 128 + w * 2;
    float v = 0.25f * (xp[0] + xp[1] + xp[128] + xp[129]) * coff[b * 128 + c0 + ci];
    tile[ci][w] = (_Float16)v;
  }
  __syncthreads();
  #pragma unroll
  for (int it = 0; it < 4; ++it) {
    int cl = t & 15;
    int w  = (t >> 4) + it * 16;
    t0h[(((long)b * 64 + h) * 64 + w) * 128 + c0 + cl] = tile[cl][w];
  }
}

// NCHW f32 -> padded NHWC f16 with coff scale (interior only; border pre-zeroed).
// grid (8 ctiles, 128 h, 32 b), 256 threads
__global__ __launch_bounds__(256) void prep_x_kernel(
    const float* __restrict__ x, const float* __restrict__ coff,
    _Float16* __restrict__ xh)
{
  __shared__ _Float16 tile[16][128];
  int c0 = blockIdx.x * 16, h = blockIdx.y, b = blockIdx.z;
  int t = threadIdx.x;
  #pragma unroll
  for (int it = 0; it < 8; ++it) {
    int ci = it * 2 + (t >> 7);
    int w  = t & 127;
    float v = x[(((long)b * 128 + c0 + ci) * 128 + h) * 128 + w] * coff[b * 128 + c0 + ci];
    tile[ci][w] = (_Float16)v;
  }
  __syncthreads();
  #pragma unroll
  for (int it = 0; it < 8; ++it) {
    int cl = t & 15;
    int w  = (t >> 4) + it * 16;
    xh[(((long)b * 130 + h + 1) * 130 + (w + 1)) * 128 + c0 + cl] = tile[cl][w];
  }
}

// Static weights [Co][Ci][3][3] f32 -> fragment-shuffled f16 [cotile][ks][lane][16]
// with tap-major K (k = tap*Ci + ci).
__global__ void wshuf_kernel(const float* __restrict__ w, _Float16* __restrict__ dst,
                             int Ci, int ciShift, long total)
{
  long idx = (long)blockIdx.x * 256 + threadIdx.x;
  if (idx >= total) return;
  int nk = (Ci * 9) >> 5;
  int s = (int)(idx & 15); long r = idx >> 4;
  int lane = (int)(r & 31); r >>= 5;
  int ks = (int)(r % nk); int ntile = (int)(r / nk);
  int k = (ks << 5) + frag_k(lane, s);
  int tap = k >> ciShift, ci = k & (Ci - 1);
  int co = ntile * 16 + (lane & 15);
  dst[idx] = (_Float16)w[((long)co * Ci + ci) * 9 + tap];
}

// Global average pool over NHWC f16: out[b*C+c] (f32)
__global__ void gap_h_kernel(const _Float16* __restrict__ in, float* __restrict__ out,
                             int C, int npos, int total)
{
  int idx = blockIdx.x * 256 + threadIdx.x;
  if (idx >= total) return;
  int b = idx / C, c = idx - b * C;
  float s = 0.f;
  for (int p = 0; p < npos; ++p) s += (float)in[((long)b * npos + p) * C + c];
  out[idx] = s / (float)npos;
}

// kernel_att[b,e] = lrelu(a[b,:], 0.01) @ att_fw[e,:] + att_fb[e]
__global__ void att_kernel(const float* __restrict__ a,
                           const float* __restrict__ fw, const float* __restrict__ fb,
                           float* __restrict__ katt)
{
  int t = threadIdx.x;
  if (t >= 128) return;
  int b = t >> 2, e = t & 3;
  float s = fb[e];
  for (int c = 0; c < 128; ++c) {
    float v = a[b * 128 + c];
    v = (v >= 0.f) ? v : 0.01f * v;
    s += v * fw[e * 128 + c];
  }
  katt[b * 4 + e] = s;
}

// Per-sample dynamic weights, written directly in fragment-shuffled order:
// [b][ntile 0..3][ks 0..35][lane][16];  w[b,o,ci,tap] = sum_e katt*agg*expert
__global__ void wdyn_kernel(const float* __restrict__ katt,
                            const float* __restrict__ agg,
                            const float* __restrict__ ew,
                            _Float16* __restrict__ wd)
{
  long idx = (long)blockIdx.x * 256 + threadIdx.x;  // 32 * 73728
  if (idx >= 32L * 73728) return;
  int s = (int)(idx & 15); long r = idx >> 4;
  int lane = (int)(r & 31); r >>= 5;
  int ks = (int)(r % 36); r /= 36;
  int ntile = (int)(r & 3); int b = (int)(r >> 2);
  int k = (ks << 5) + frag_k(lane, s);
  int tap = k >> 7, ci = k & 127;
  int o = ntile * 16 + (lane & 15);
  float v = 0.f;
  #pragma unroll
  for (int e = 0; e < 4; ++e)
    v += katt[b * 4 + e] * agg[((long)e * 32 + b) * 128 + ci]
       * ew[(((long)e * 64 + o) * 128 + ci) * 9 + tap];
  wd[idx] = (_Float16)v;
}

extern "C" void kernel_launch(void* const* d_in, const int* in_sizes, int n_in,
                              void* d_out, int out_size, void* d_ws, size_t ws_size,
                              hipStream_t stream) {
  (void)in_sizes; (void)n_in; (void)out_size; (void)ws_size;
  const float* x      = (const float*)d_in[0];
  const float* wm     = (const float*)d_in[1];
  const float* wm_w1  = (const float*)d_in[2];
  const float* wm_b1  = (const float*)d_in[3];
  const float* wm_w2  = (const float*)d_in[4];
  const float* wm_b2  = (const float*)d_in[5];
  const float* tr_w1  = (const float*)d_in[6];
  const float* tr_b1  = (const float*)d_in[7];
  const float* tr_w2  = (const float*)d_in[8];
  const float* tr_b2  = (const float*)d_in[9];
  const float* tr_w3  = (const float*)d_in[10];
  const float* tr_b3  = (const float*)d_in[11];
  const float* h_w[4] = {(const float*)d_in[12], (const float*)d_in[14],
                         (const float*)d_in[16], (const float*)d_in[18]};
  const float* h_b[4] = {(const float*)d_in[13], (const float*)d_in[15],
                         (const float*)d_in[17], (const float*)d_in[19]};
  const float* att_cw = (const float*)d_in[20];
  const float* att_cb = (const float*)d_in[21];
  const float* att_fw = (const float*)d_in[22];
  const float* att_fb = (const float*)d_in[23];
  const float* ew     = (const float*)d_in[24];

  // ---- workspace (256B-aligned bump; big buffers share one arena) ----
  char* ws = (char*)d_ws;
  size_t off = 0;
  auto bump = [&](size_t bytes) -> char* {
    char* p = ws + off;
    off += (bytes + 255) & ~(size_t)255;
    return p;
  };
  float*    coff  = (float*)   bump(32 * 128 * 4);
  float*    katt  = (float*)   bump(32 * 4 * 4);
  float*    agg   = (float*)   bump(4 * 32 * 128 * 4);
  float*    atta  = (float*)   bump(32 * 128 * 4);
  _Float16* wdynh = (_Float16*)bump(32L * 73728 * 2);      // 4.7 MB
  _Float16* tr1h  = (_Float16*)bump(64L * 1152 * 2);
  _Float16* tr2h  = (_Float16*)bump(64L * 576 * 2);
  _Float16* tr3h  = (_Float16*)bump(64L * 576 * 2);
  _Float16* hh[5];
  for (int e = 0; e < 5; ++e) hh[e] = (_Float16*)bump(128L * 576 * 2);
  _Float16* houth[5];
  for (int e = 0; e < 5; ++e) houth[e] = (_Float16*)bump(32L * 36 * 128 * 2);
  _Float16* t3h = (_Float16*)bump(32L * 196 * 64 * 2);
  // arena: t0h/t1h/t2h live only during the tower; xh (138.4MB) reuses it later
  char* arena = bump(32L * 130 * 130 * 128 * 2);
  _Float16* t0h = (_Float16*)arena;                         // 33.6 MB
  _Float16* t1h = (_Float16*)(arena + ((32L*64*64*128*2 + 255) & ~255L));
  _Float16* t2h = (_Float16*)((char*)t1h + ((32L*3844*64*2 + 255) & ~255L));
  _Float16* xh  = (_Float16*)arena;                         // overwrites tower bufs

  // 1) watermark MLP -> channel modulation coefficients
  mlp_coff_kernel<<<1, 128, 0, stream>>>(wm, wm_w1, wm_b1, wm_w2, wm_b2, coff);

  // 2) shuffle static conv weights into fragment order (f16, tap-major K)
  wshuf_kernel<<<288, 256, 0, stream>>>(tr_w1, tr1h, 128, 7, 64L * 1152);
  wshuf_kernel<<<144, 256, 0, stream>>>(tr_w2, tr2h, 64, 6, 64L * 576);
  wshuf_kernel<<<144, 256, 0, stream>>>(tr_w3, tr3h, 64, 6, 64L * 576);
  for (int e = 0; e < 4; ++e)
    wshuf_kernel<<<288, 256, 0, stream>>>(h_w[e], hh[e], 64, 6, 128L * 576);
  wshuf_kernel<<<288, 256, 0, stream>>>(att_cw, hh[4], 64, 6, 128L * 576);

  // 3) fused coff-scale + avgpool -> t0h NHWC f16 [32,64,64,128]
  scale_pool_kernel<<<dim3(8, 64, 32), 256, 0, stream>>>(x, coff, t0h);

  // 4) feature tower (lrelu 0.01), NHWC f16 chain
  conv_wmma_kernel<<<dim3(31, 1, 32), 256, 0, stream>>>(
      t0h, tr1h, tr_b1, t1h, 128, 7, 64, 64, 64, 62, 62, 1, 0L, 0.01f, 0);
  conv_wmma_kernel<<<dim3(8, 1, 32), 256, 0, stream>>>(
      t1h, tr2h, tr_b2, t2h, 64, 6, 62, 62, 64, 30, 30, 2, 0L, 0.01f, 0);
  conv_wmma_kernel<<<dim3(2, 1, 32), 256, 0, stream>>>(
      t2h, tr3h, tr_b3, t3h, 64, 6, 30, 30, 64, 14, 14, 2, 0L, 0.01f, 0);

  // 5) 4 coefficient heads + attention conv (bias, no activation)
  for (int e = 0; e < 4; ++e)
    conv_wmma_kernel<<<dim3(1, 2, 32), 256, 0, stream>>>(
        t3h, hh[e], h_b[e], houth[e], 64, 6, 14, 14, 128, 6, 6, 2, 0L, 1.0f, 0);
  conv_wmma_kernel<<<dim3(1, 2, 32), 256, 0, stream>>>(
      t3h, hh[4], att_cb, houth[4], 64, 6, 14, 14, 128, 6, 6, 2, 0L, 1.0f, 0);

  // 6) GAPs -> agg[e][32][128], attention vector
  for (int e = 0; e < 4; ++e)
    gap_h_kernel<<<16, 256, 0, stream>>>(houth[e], agg + (long)e * 32 * 128, 128, 36, 32 * 128);
  gap_h_kernel<<<16, 256, 0, stream>>>(houth[4], atta, 128, 36, 32 * 128);

  // 7) attention over 4 experts
  att_kernel<<<1, 128, 0, stream>>>(atta, att_fw, att_fb, katt);

  // 8) per-sample dynamic weights, fragment-shuffled f16
  wdyn_kernel<<<(int)((32L * 73728 + 255) / 256), 256, 0, stream>>>(katt, agg, ew, wdynh);

  // 9) padded NHWC f16 x (tower buffers dead -> arena reuse)
  hipMemsetAsync(xh, 0, 32L * 130 * 130 * 128 * 2, stream);
  prep_x_kernel<<<dim3(8, 128, 32), 256, 0, stream>>>(x, coff, xh);

  // 10) final per-sample conv 128->64, k3 s1 p1, f32 NCHW into d_out
  conv_wmma_kernel<<<dim3(128, 1, 32), 256, 0, stream>>>(
      xh, wdynh, nullptr, d_out, 128, 7, 130, 130, 64, 128, 128, 1, 73728L, 1.0f, 1);
}
